// NDA_pam_7550552507246
// MI455X (gfx1250) — matrix operand order repaired and checked by reading, not compiled, not run on hardware
//
#include <hip/hip_runtime.h>
#include <hip/hip_bf16.h>

// ---------------------------------------------------------------------------
// NDA / PAM attention pipeline for MI455X (gfx1250, wave32, WMMA bf16).
//   B=1, C=8192, H=W=32 -> N=1024, C8=1024
// All big GEMMs run on v_wmma_f32_16x16x32_bf16 with f32 accumulation,
// register-staged double-buffered LDS pipeline (1 barrier / k-step).
// ---------------------------------------------------------------------------

typedef __attribute__((ext_vector_type(16))) __bf16 bf16x16;
typedef __attribute__((ext_vector_type(8)))  __bf16 bf16x8;
typedef __attribute__((ext_vector_type(4)))  __bf16 bf16x4;
typedef __attribute__((ext_vector_type(8)))  float  f32x8;

static constexpr int BK  = 32;   // k tile == WMMA K
static constexpr int LDT = 40;   // padded LDS row stride (bf16): 80B, 16B-aligned,
                                 // distinct bank groups for frag b128 reads

template <typename T> struct AVecSel { using type = bf16x4; };
template <> struct AVecSel<float>    { using type = float4; };

// ---------------------------------------------------------------------------
// f32 -> bf16 convert (x staging; xb stays L2-resident at 16MB)
// ---------------------------------------------------------------------------
__global__ __launch_bounds__(256) void cvt_f32_bf16(const float* __restrict__ X,
                                                    __bf16* __restrict__ Y, int total4) {
  int i = blockIdx.x * 256 + threadIdx.x;
  if (i < total4) {
    float4 f = ((const float4*)X)[i];
    bf16x4 o;
    o[0] = (__bf16)f.x; o[1] = (__bf16)f.y; o[2] = (__bf16)f.z; o[3] = (__bf16)f.w;
    ((bf16x4*)Y)[i] = o;
  }
}

// ---------------------------------------------------------------------------
// Generic WMMA GEMM:  Y = epilogue( op(A) @ op(B) ),  TBM x TBN tile.
//   AT: float (converted to bf16 while staging) or __bf16
//   TA: A stored (K x M) row-major            (cov: qc^T)
//   TB: B stored (N x K) row-major, bf16      (final: S^T)
//   EPI 0: y = acc*rowscale[m] + rowshift[m]  -> f32   (q, k)
//   EPI 1: y = acc*rowscale[m] + rowshift[m]  -> bf16  (v)
//   EPI 2: y = acc*colscale[n]                -> f32   (ppmcc)
//   EPI 3: y = colscale[0]*acc + xres[m*N+n]  -> f32   (final)
// 256 threads = 8 waves arranged 2(M) x 4(N); each wave owns
// (TBM/2) rows x (TBN/4) cols = (TBM/32)x(TBN/64) WMMA tiles.
// All dims are multiples of the tile (1024/8192) -> no bounds checks.
// ---------------------------------------------------------------------------
template <typename AT, bool TA, bool TB, int EPI, int TBM, int TBN>
__global__ __launch_bounds__(256) void gemm_wmma(
    const AT* __restrict__ A, const __bf16* __restrict__ B,
    float* __restrict__ Yf, __bf16* __restrict__ Yb,
    const float* __restrict__ rowscale, const float* __restrict__ rowshift,
    const float* __restrict__ colscale, const float* __restrict__ xres,
    int M, int N, int K) {
  constexpr int MT    = TBM / 32;   // m-subtiles per wave
  constexpr int NT    = TBN / 64;   // n-subtiles per wave
  constexpr int APASS = TBM / 32;   // staging chunks (4 elems) per thread, A
  constexpr int BPASS = TBN / 32;   // staging chunks per thread, B
  using AVec = typename AVecSel<AT>::type;

  __shared__ __bf16 Asm[2][TBM * LDT];
  __shared__ __bf16 Bsm[2][TBN * LDT];

  const int tid  = threadIdx.x;
  const int lane = tid & 31;
  const int wv   = tid >> 5;
  const int wm   = wv >> 2;              // 0..1
  const int wn   = wv & 3;               // 0..3
  const int m0   = blockIdx.y * TBM;
  const int n0   = blockIdx.x * TBN;
  const int lr   = lane & 15;            // fragment row/col within 16
  const int kh   = lane >> 4;            // K-half select

  f32x8 acc[MT][NT] = {};
  AVec   aReg[APASS];
  bf16x4 bReg[BPASS];

  // ---- global tile -> registers (issued ahead of compute) ----
  auto load_regs = [&](int k0) {
    if constexpr (!TA) {
      #pragma unroll
      for (int p = 0; p < APASS; ++p) {
        int c = p * 256 + tid;
        int m = c >> 3, kv = (c & 7) << 2;
        aReg[p] = *(const AVec*)(A + (size_t)(m0 + m) * K + (k0 + kv));
      }
    } else {
      #pragma unroll
      for (int p = 0; p < APASS; ++p) {
        int c  = p * 256 + tid;
        int kk = c / (TBM / 4), mv = (c % (TBM / 4)) << 2;
        aReg[p] = *(const bf16x4*)(A + (size_t)(k0 + kk) * M + (m0 + mv));
      }
    }
    if constexpr (!TB) {
      #pragma unroll
      for (int p = 0; p < BPASS; ++p) {
        int c  = p * 256 + tid;
        int kk = c / (TBN / 4), nv = (c % (TBN / 4)) << 2;
        bReg[p] = *(const bf16x4*)(B + (size_t)(k0 + kk) * N + (n0 + nv));
      }
    } else {
      #pragma unroll
      for (int p = 0; p < BPASS; ++p) {
        int c   = p * 256 + tid;
        int col = c >> 3, kv = (c & 7) << 2;
        bReg[p] = *(const bf16x4*)(B + (size_t)(n0 + col) * K + (k0 + kv));
      }
    }
  };

  // ---- registers -> LDS buffer (bf16, [row][k] layout) ----
  auto store_lds = [&](int buf) {
    __bf16* As = Asm[buf];
    __bf16* Bs = Bsm[buf];
    if constexpr (!TA) {
      #pragma unroll
      for (int p = 0; p < APASS; ++p) {
        int c = p * 256 + tid;
        int m = c >> 3, kv = (c & 7) << 2;
        bf16x4 o;
        if constexpr (sizeof(AT) == 4) {
          float4 f = aReg[p];
          o[0] = (__bf16)f.x; o[1] = (__bf16)f.y; o[2] = (__bf16)f.z; o[3] = (__bf16)f.w;
        } else {
          o = aReg[p];
        }
        *(bf16x4*)&As[m * LDT + kv] = o;
      }
    } else {  // transpose while storing
      #pragma unroll
      for (int p = 0; p < APASS; ++p) {
        int c  = p * 256 + tid;
        int kk = c / (TBM / 4), mv = (c % (TBM / 4)) << 2;
        bf16x4 f = aReg[p];
        As[(mv + 0) * LDT + kk] = f[0];
        As[(mv + 1) * LDT + kk] = f[1];
        As[(mv + 2) * LDT + kk] = f[2];
        As[(mv + 3) * LDT + kk] = f[3];
      }
    }
    if constexpr (!TB) {  // transpose while storing
      #pragma unroll
      for (int p = 0; p < BPASS; ++p) {
        int c  = p * 256 + tid;
        int kk = c / (TBN / 4), nv = (c % (TBN / 4)) << 2;
        bf16x4 f = bReg[p];
        Bs[(nv + 0) * LDT + kk] = f[0];
        Bs[(nv + 1) * LDT + kk] = f[1];
        Bs[(nv + 2) * LDT + kk] = f[2];
        Bs[(nv + 3) * LDT + kk] = f[3];
      }
    } else {  // already [col][k] oriented
      #pragma unroll
      for (int p = 0; p < BPASS; ++p) {
        int c   = p * 256 + tid;
        int col = c >> 3, kv = (c & 7) << 2;
        *(bf16x4*)&Bs[col * LDT + kv] = bReg[p];
      }
    }
  };

  const int iters = K / BK;
  load_regs(0);
  store_lds(0);

  for (int it = 0; it < iters; ++it) {
    __syncthreads();                       // buf[it&1] ready for all waves
    const int  cur  = it & 1;
    const bool more = (it + 1) < iters;
    if (more) load_regs((it + 1) * BK);    // global loads in flight during WMMA

    const __bf16* As = Asm[cur];
    const __bf16* Bs = Bsm[cur];
    // B frags (ISA layout: lane lr = col; K = kh*8 + {0..7}, then +16)
    bf16x16 bfrag[NT];
    #pragma unroll
    for (int nt = 0; nt < NT; ++nt) {
      int col = wn * (TBN / 4) + nt * 16 + lr;
      bf16x8 lo = *(const bf16x8*)&Bs[col * LDT + kh * 8];
      bf16x8 hi = *(const bf16x8*)&Bs[col * LDT + 16 + kh * 8];
      #pragma unroll
      for (int i = 0; i < 8; ++i) { bfrag[nt][i] = lo[i]; bfrag[nt][8 + i] = hi[i]; }
    }
    #pragma unroll
    for (int mt = 0; mt < MT; ++mt) {
      // A frag (ISA layout: lane lr = row; K = kh*8 + {0..7}, then +16)
      int row = wm * (TBM / 2) + mt * 16 + lr;
      bf16x8 lo = *(const bf16x8*)&As[row * LDT + kh * 8];
      bf16x8 hi = *(const bf16x8*)&As[row * LDT + 16 + kh * 8];
      bf16x16 af;
      #pragma unroll
      for (int i = 0; i < 8; ++i) { af[i] = lo[i]; af[8 + i] = hi[i]; }
      #pragma unroll
      for (int nt = 0; nt < NT; ++nt) {
        acc[mt][nt] = __builtin_amdgcn_wmma_f32_16x16x32_bf16(
            false, af, false, bfrag[nt], (short)0, acc[mt][nt], false, false);
      }
    }
    if (more) store_lds(cur ^ 1);          // next barrier publishes it
  }

  // ---- epilogue (C/D layout: VGPR v -> row kh*8+v, col = lr) ----
  #pragma unroll
  for (int mt = 0; mt < MT; ++mt) {
    #pragma unroll
    for (int nt = 0; nt < NT; ++nt) {
      const int gc  = n0 + wn * (TBN / 4) + nt * 16 + lr;
      const int grb = m0 + wm * (TBM / 2) + mt * 16 + kh * 8;
      if constexpr (EPI == 0 || EPI == 1) {
        #pragma unroll
        for (int v = 0; v < 8; ++v) {
          int gr = grb + v;
          float y = acc[mt][nt][v] * rowscale[gr] + rowshift[gr];
          if constexpr (EPI == 0) Yf[(size_t)gr * N + gc] = y;
          else                    Yb[(size_t)gr * N + gc] = (__bf16)y;
        }
      } else if constexpr (EPI == 2) {
        float cs = colscale[gc];
        #pragma unroll
        for (int v = 0; v < 8; ++v)
          Yf[(size_t)(grb + v) * N + gc] = acc[mt][nt][v] * cs;
      } else {
        float g = colscale[0];
        #pragma unroll
        for (int v = 0; v < 8; ++v) {
          size_t idx = (size_t)(grb + v) * N + gc;
          Yf[idx] = g * acc[mt][nt][v] + xres[idx];
        }
      }
    }
  }
}

// ---------------------------------------------------------------------------
// Per-row mean/std of q and k (ddof=0), centered bf16 outputs, and the fused
// PPMCC column scale: scl[r] = 1 / ((N-1) * std_q[r] * std_k[r]).
// ---------------------------------------------------------------------------
__global__ __launch_bounds__(256) void rowstats(
    const float* __restrict__ Q, const float* __restrict__ Km,
    __bf16* __restrict__ Qc, __bf16* __restrict__ Kc,
    float* __restrict__ scl, int N) {
  __shared__ float4 rbuf[256];
  __shared__ float  mq, mk;
  const int r = blockIdx.x, t = threadIdx.x;
  const float* q = Q  + (size_t)r * N;
  const float* k = Km + (size_t)r * N;
  float s1q = 0.f, s2q = 0.f, s1k = 0.f, s2k = 0.f;
  for (int j = t; j < N; j += 256) {
    float a = q[j]; s1q += a; s2q += a * a;
    float b = k[j]; s1k += b; s2k += b * b;
  }
  rbuf[t] = make_float4(s1q, s2q, s1k, s2k);
  __syncthreads();
  for (int s = 128; s > 0; s >>= 1) {
    if (t < s) {
      float4 a = rbuf[t], b = rbuf[t + s];
      rbuf[t] = make_float4(a.x + b.x, a.y + b.y, a.z + b.z, a.w + b.w);
    }
    __syncthreads();
  }
  if (t == 0) {
    float4 s  = rbuf[0];
    float rn  = 1.0f / (float)N;
    float meanq = s.x * rn, meank = s.z * rn;
    float vq = fmaxf(s.y * rn - meanq * meanq, 0.f);
    float vk = fmaxf(s.w * rn - meank * meank, 0.f);
    scl[r] = 1.0f / ((float)(N - 1) * sqrtf(vq) * sqrtf(vk));
    mq = meanq; mk = meank;
  }
  __syncthreads();
  float meanq = mq, meank = mk;
  for (int j = t; j < N; j += 256) {
    Qc[(size_t)r * N + j] = (__bf16)(q[j] - meanq);
    Kc[(size_t)r * N + j] = (__bf16)(k[j] - meank);
  }
}

// ---------------------------------------------------------------------------
// Row softmax of ppmcc (1024 x 1024), one pass, values in registers.
// Output S bf16 row-major (m x n); final GEMM consumes it as B-transposed.
// ---------------------------------------------------------------------------
__global__ __launch_bounds__(256) void softmax_rows(const float* __restrict__ P,
                                                    __bf16* __restrict__ S, int N) {
  __shared__ float red[256];
  const int r = blockIdx.x, t = threadIdx.x;
  const float* p = P + (size_t)r * N;
  float vals[4];
  float mx = -3.402823466e38f;
  #pragma unroll
  for (int i = 0; i < 4; ++i) { vals[i] = p[t + i * 256]; mx = fmaxf(mx, vals[i]); }
  red[t] = mx; __syncthreads();
  for (int s = 128; s > 0; s >>= 1) { if (t < s) red[t] = fmaxf(red[t], red[t + s]); __syncthreads(); }
  mx = red[0]; __syncthreads();
  float sum = 0.f;
  #pragma unroll
  for (int i = 0; i < 4; ++i) { vals[i] = __expf(vals[i] - mx); sum += vals[i]; }
  red[t] = sum; __syncthreads();
  for (int s = 128; s > 0; s >>= 1) { if (t < s) red[t] += red[t + s]; __syncthreads(); }
  float rinv = 1.0f / red[0];
  #pragma unroll
  for (int i = 0; i < 4; ++i)
    S[(size_t)r * N + t + i * 256] = (__bf16)(vals[i] * rinv);
}

// ---------------------------------------------------------------------------
// Launcher
// ---------------------------------------------------------------------------
extern "C" void kernel_launch(void* const* d_in, const int* in_sizes, int n_in,
                              void* d_out, int out_size, void* d_ws, size_t ws_size,
                              hipStream_t stream) {
  constexpr int C  = 8192;
  constexpr int N  = 1024;   // H*W
  constexpr int C8 = 1024;

  const float* x       = (const float*)d_in[0];
  const float* Wq      = (const float*)d_in[1];
  const float* Wk      = (const float*)d_in[2];
  const float* Wv      = (const float*)d_in[3];
  const float* q_scale = (const float*)d_in[4];
  const float* q_shift = (const float*)d_in[5];
  const float* k_scale = (const float*)d_in[6];
  const float* k_shift = (const float*)d_in[7];
  const float* v_scale = (const float*)d_in[8];
  const float* v_shift = (const float*)d_in[9];
  const float* gamma   = (const float*)d_in[10];
  float* out = (float*)d_out;

  // workspace carve-up (256B aligned), ~50 MB total
  char* ws = (char*)d_ws;
  size_t off = 0;
  auto carve = [&](size_t bytes) { void* p = ws + off; off = (off + bytes + 255) & ~(size_t)255; return p; };
  __bf16* xb  = (__bf16*)carve((size_t)C * N * 2);    // x in bf16           16 MB
  float*  qf  = (float*) carve((size_t)C8 * N * 4);   // q (f32)              4 MB
  float*  kf  = (float*) carve((size_t)C8 * N * 4);   // k (f32)              4 MB
  __bf16* vb  = (__bf16*)carve((size_t)C * N * 2);    // v (bf16)            16 MB
  __bf16* qc  = (__bf16*)carve((size_t)C8 * N * 2);   // centered q           2 MB
  __bf16* kc  = (__bf16*)carve((size_t)C8 * N * 2);   // centered k           2 MB
  float*  scl = (float*) carve((size_t)C8 * 4);       // ppmcc col scale
  float*  pp  = (float*) carve((size_t)N * N * 4);    // ppmcc                4 MB
  __bf16* sb  = (__bf16*)carve((size_t)N * N * 2);    // softmax probs        2 MB
  (void)ws_size; (void)in_sizes; (void)n_in; (void)out_size;

  // 1) x -> bf16 (L2-resident operand for all three projection GEMMs)
  cvt_f32_bf16<<<(C * N / 4 + 255) / 256, 256, 0, stream>>>(x, xb, C * N / 4);

  // 2) q = (Wq @ x) * qs + qh   [f32 out]   128x128 tiles
  gemm_wmma<float, false, false, 0, 128, 128><<<dim3(N / 128, C8 / 128), 256, 0, stream>>>(
      Wq, xb, qf, nullptr, q_scale, q_shift, nullptr, nullptr, C8, N, C);
  // 3) k = (Wk @ x) * ks + kh   [f32 out]
  gemm_wmma<float, false, false, 0, 128, 128><<<dim3(N / 128, C8 / 128), 256, 0, stream>>>(
      Wk, xb, kf, nullptr, k_scale, k_shift, nullptr, nullptr, C8, N, C);
  // 4) v = (Wv @ x) * vs + vh   [bf16 out]  128x256 tiles: halves Wv HBM re-reads
  gemm_wmma<float, false, false, 1, 128, 256><<<dim3(N / 256, C / 128), 256, 0, stream>>>(
      Wv, xb, nullptr, vb, v_scale, v_shift, nullptr, nullptr, C, N, C);

  // 5) row stats + centering + fused ppmcc scale
  rowstats<<<C8, 256, 0, stream>>>(qf, kf, qc, kc, scl, N);

  // 6) ppmcc = (qc^T @ kc) * scl[col]
  gemm_wmma<__bf16, true, false, 2, 128, 128><<<dim3(N / 128, N / 128), 256, 0, stream>>>(
      qc, kc, pp, nullptr, nullptr, nullptr, scl, nullptr, N, N, C8);

  // 7) S = softmax_rows(ppmcc)  [bf16, row-major (m x n)]
  softmax_rows<<<N, 256, 0, stream>>>(pp, sb, N);

  // 8) out = gamma * (v @ S^T) + x        128x256 tiles
  gemm_wmma<__bf16, false, true, 3, 128, 256><<<dim3(N / 256, C / 128), 256, 0, stream>>>(
      vb, sb, out, nullptr, nullptr, nullptr, gamma, x, C, N, N);
}
